// RENLayer_16922171146740
// MI455X (gfx1250) — compile-verified, matrix operand order checked by reading, not run on hardware
//
#include <hip/hip_runtime.h>

// ---------------------------------------------------------------------------
// REN layer (forward) for MI455X / gfx1250.
// v_wmma_f32_16x16x32_bf16 + double-buffered LDS staging via
// GLOBAL_LOAD_ASYNC_TO_LDS_B128 (ASYNCcnt) when available.
//
//   b      = x@C1 + u@D12                      (f32)
//   w_{t+1}= tanh(b + w_t@D11)   x30           (bf16 ping-pong, L2-resident)
//   x_next = x@A + w@B1 + u@B2   -> d_out[0:8192*1024]
//   y      = x@C2 + w@D21 + u@D22-> d_out[8192*1024:]
//
// Weights pre-transposed to Wt[n][k] bf16 so A and B tiles stage identically.
// ---------------------------------------------------------------------------

typedef __attribute__((ext_vector_type(16))) __bf16 v16bf;
typedef __attribute__((ext_vector_type(8)))  __bf16 v8bf;
typedef __attribute__((ext_vector_type(8)))  float  v8f;

#define DEV static __device__ __forceinline__

#if defined(__has_builtin)
#if __has_builtin(__builtin_amdgcn_global_load_async_to_lds_b128) && \
    __has_builtin(__builtin_amdgcn_s_wait_asynccnt)
#define HAVE_ASYNC_LDS 1
#endif
#endif

// LDS tile geometry: 128 rows x 32 bf16 cols, pitch 40 elements (80B) so the
// 16 lanes of a fragment read hit 16 disjoint 4-bank groups (conflict-free).
#define TILE_PITCH 40
#define TILE_ELEMS (128 * TILE_PITCH)

// copy 16 bytes global -> LDS
DEV void copy16(const __bf16* gsrc, __bf16* ldst) {
#ifdef HAVE_ASYNC_LDS
  // builtin signature (per hipcc diagnostic): (int4 AS1*, int4 AS3*, Ii, Ii)
  typedef int v4i_t __attribute__((vector_size(4 * sizeof(int))));
  typedef __attribute__((address_space(1))) v4i_t* gptr_t;
  typedef __attribute__((address_space(3))) v4i_t* lptr_t;
  __builtin_amdgcn_global_load_async_to_lds_b128(
      (gptr_t)(unsigned long long)gsrc,
      (lptr_t)(unsigned)(unsigned long long)ldst, 0, 0);
#else
  *(v8bf*)ldst = *(const v8bf*)gsrc;
#endif
}

// stage one 128x32 bf16 tile (rows tile_row0.., cols kk..kk+31) into LDS.
// 256 threads: thread t handles row t>>1, 16-col half (t&1). 2 x b128 each.
DEV void stage_tile(const __bf16* __restrict__ g, int ld, int tile_row0, int kk,
                    __bf16* lds, int tid) {
  const int row  = tid >> 1;
  const int colc = (tid & 1) << 4;  // 0 or 16
  const __bf16* src = g + (size_t)(tile_row0 + row) * ld + kk + colc;
  __bf16* dst = lds + row * TILE_PITCH + colc;
  copy16(src, dst);
  copy16(src + 8, dst + 8);
}

// read one 16x32 WMMA fragment from an LDS tile (CDNA5 16-bit A layout):
// lane L: row = row0 + (L&15); elems 0..7 = K[half*8..+7], 8..15 = +16.
DEV v16bf lds_frag(const __bf16* lds, int row0, int lane) {
  const int r  = row0 + (lane & 15);
  const int kb = (lane >> 4) << 3;
  const __bf16* p = lds + r * TILE_PITCH + kb;
  v8bf lo = *(const v8bf*)p;
  v8bf hi = *(const v8bf*)(p + 16);
  return __builtin_shufflevector(lo, hi, 0, 1, 2, 3, 4, 5, 6, 7, 8, 9, 10, 11,
                                 12, 13, 14, 15);
}

// accumulate one (A[MxK], Wt[NxK]) pair with double-buffered LDS staging.
DEV void gemm_pair(const __bf16* __restrict__ A, const __bf16* __restrict__ Wt,
                   int K, int m_blk, int n_blk, int tid, int lane, int wm,
                   int wn, __bf16* sA, __bf16* sB, v8f (&acc)[4][2]) {
  const int S = K >> 5;  // K-steps of 32
  stage_tile(A, K, m_blk, 0, sA, tid);
  stage_tile(Wt, K, n_blk, 0, sB, tid);
  for (int s = 0; s < S; ++s) {
    const int cur = s & 1;
    const int nxt = cur ^ 1;
    const bool more = (s + 1) < S;
    if (more) {
      stage_tile(A, K, m_blk, (s + 1) << 5, sA + nxt * TILE_ELEMS, tid);
      stage_tile(Wt, K, n_blk, (s + 1) << 5, sB + nxt * TILE_ELEMS, tid);
    }
#ifdef HAVE_ASYNC_LDS
    if (more)
      __builtin_amdgcn_s_wait_asynccnt(4);  // batch s done, s+1 in flight
    else
      __builtin_amdgcn_s_wait_asynccnt(0);
#endif
    __syncthreads();

    v16bf af[4];
    v16bf wf[2];
#pragma unroll
    for (int t = 0; t < 4; ++t)
      af[t] = lds_frag(sA + cur * TILE_ELEMS, wm * 64 + t * 16, lane);
#pragma unroll
    for (int t = 0; t < 2; ++t)
      wf[t] = lds_frag(sB + cur * TILE_ELEMS, wn * 32 + t * 16, lane);
#pragma unroll
    for (int tm = 0; tm < 4; ++tm)
#pragma unroll
      for (int tn = 0; tn < 2; ++tn)
        acc[tm][tn] = __builtin_amdgcn_wmma_f32_16x16x32_bf16(
            false, af[tm], false, wf[tn], (short)0, acc[tm][tn], false, false);

    __syncthreads();  // all waves done with 'cur' before it is restaged
  }
}

// --- fused WMMA GEMM: out[M,N] = act(bias + sum_p Ap @ Wp^T), <=3 pairs ----
// block = 256 threads (8 wave32), block tile 128x128, wave tile 64x32.
__global__ __launch_bounds__(256) void k_wmma_gemm(
    const __bf16* __restrict__ A0, const __bf16* __restrict__ W0, int K0,
    const __bf16* __restrict__ A1, const __bf16* __restrict__ W1, int K1,
    const __bf16* __restrict__ A2, const __bf16* __restrict__ W2, int K2,
    const float* __restrict__ bias, float* __restrict__ outF,
    __bf16* __restrict__ outB, int M, int N, int act) {
  __shared__ __bf16 sA[2 * TILE_ELEMS];
  __shared__ __bf16 sB[2 * TILE_ELEMS];

  const int tid  = threadIdx.x;
  const int lane = tid & 31;
  const int wave = tid >> 5;  // 0..7
  const int wm = wave & 1;    // 2 wave rows   (64 rows each)
  const int wn = wave >> 1;   // 4 wave cols   (32 cols each)
  const int m_blk = blockIdx.x * 128;
  const int n_blk = blockIdx.y * 128;

  v8f acc[4][2];
#pragma unroll
  for (int tm = 0; tm < 4; ++tm)
#pragma unroll
    for (int tn = 0; tn < 2; ++tn)
#pragma unroll
      for (int e = 0; e < 8; ++e) acc[tm][tn][e] = 0.0f;

  gemm_pair(A0, W0, K0, m_blk, n_blk, tid, lane, wm, wn, sA, sB, acc);
  if (K1) gemm_pair(A1, W1, K1, m_blk, n_blk, tid, lane, wm, wn, sA, sB, acc);
  if (K2) gemm_pair(A2, W2, K2, m_blk, n_blk, tid, lane, wm, wn, sA, sB, acc);

  // epilogue: C/D layout -> row = e + 8*(lane>>4), col = lane&15 per tile
  const int colL = lane & 15;
  const int rowO = (lane >> 4) << 3;
  const int m_base = m_blk + wm * 64;
  const int n_base = n_blk + wn * 32;
#pragma unroll
  for (int tm = 0; tm < 4; ++tm) {
#pragma unroll
    for (int tn = 0; tn < 2; ++tn) {
      const int c = n_base + tn * 16 + colL;
#pragma unroll
      for (int e = 0; e < 8; ++e) {
        const int r = m_base + tm * 16 + rowO + e;
        const size_t idx = (size_t)r * N + c;
        float v = acc[tm][tn][e];
        if (bias) v += bias[idx];
        if (act) v = tanhf(v);
        if (outF)
          outF[idx] = v;
        else
          outB[idx] = (__bf16)v;
      }
    }
  }
}

// --- helpers ---------------------------------------------------------------
// src[K,N] f32 (row major) -> dst[N,K] bf16 (transposed)
__global__ void k_transpose_cvt(const float* __restrict__ src,
                                __bf16* __restrict__ dst, int K, int N) {
  size_t i = blockIdx.x * (size_t)blockDim.x + threadIdx.x;
  size_t total = (size_t)K * N;
  if (i >= total) return;
  int k = (int)(i % K);
  int n = (int)(i / K);
  dst[i] = (__bf16)src[(size_t)k * N + n];
}

__global__ void k_cvt_bf16(const float* __restrict__ src,
                           __bf16* __restrict__ dst, size_t n) {
  size_t i = blockIdx.x * (size_t)blockDim.x + threadIdx.x;
  if (i < n) dst[i] = (__bf16)src[i];
}

__global__ void k_zero_bf16(__bf16* __restrict__ dst, size_t n) {
  size_t i = blockIdx.x * (size_t)blockDim.x + threadIdx.x;
  if (i < n) dst[i] = (__bf16)0.0f;
}

// ---------------------------------------------------------------------------
extern "C" void kernel_launch(void* const* d_in, const int* in_sizes, int n_in,
                              void* d_out, int out_size, void* d_ws,
                              size_t ws_size, hipStream_t stream) {
  (void)in_sizes; (void)n_in; (void)out_size; (void)ws_size;

  const int Bb = 8192, XS = 1024, US = 512, WS = 1024, YS = 512;
  const int NITER = 30;

  const float* x   = (const float*)d_in[0];
  const float* u   = (const float*)d_in[1];
  const float* A   = (const float*)d_in[2];
  const float* B1  = (const float*)d_in[3];
  const float* B2  = (const float*)d_in[4];
  const float* C1  = (const float*)d_in[5];
  const float* D11 = (const float*)d_in[6];
  const float* D12 = (const float*)d_in[7];
  const float* C2  = (const float*)d_in[8];
  const float* D21 = (const float*)d_in[9];
  const float* D22 = (const float*)d_in[10];

  char* wsp = (char*)d_ws;
  auto alloc = [&](size_t bytes) -> void* {
    void* p = (void*)wsp;
    wsp += (bytes + 255) & ~(size_t)255;
    return p;
  };

  __bf16* C1t  = (__bf16*)alloc((size_t)XS * WS * 2);  // [WS,XS]
  __bf16* D12t = (__bf16*)alloc((size_t)US * WS * 2);  // [WS,US]
  __bf16* D11t = (__bf16*)alloc((size_t)WS * WS * 2);  // [WS,WS]
  __bf16* At   = (__bf16*)alloc((size_t)XS * XS * 2);  // [XS,XS]
  __bf16* B1t  = (__bf16*)alloc((size_t)WS * XS * 2);  // [XS,WS]
  __bf16* B2t  = (__bf16*)alloc((size_t)US * XS * 2);  // [XS,US]
  __bf16* C2t  = (__bf16*)alloc((size_t)XS * YS * 2);  // [YS,XS]
  __bf16* D21t = (__bf16*)alloc((size_t)WS * YS * 2);  // [YS,WS]
  __bf16* D22t = (__bf16*)alloc((size_t)US * YS * 2);  // [YS,US]
  __bf16* xb   = (__bf16*)alloc((size_t)Bb * XS * 2);
  __bf16* ub   = (__bf16*)alloc((size_t)Bb * US * 2);
  float*  bmat = (float*)alloc((size_t)Bb * WS * 4);
  __bf16* w0   = (__bf16*)alloc((size_t)Bb * WS * 2);
  __bf16* w1   = (__bf16*)alloc((size_t)Bb * WS * 2);

  dim3 blk(256);
  auto g1 = [](size_t n) { return dim3((unsigned)((n + 255) / 256)); };

  // 1) weights -> bf16, transposed to Wt[n][k]
  k_transpose_cvt<<<g1((size_t)XS * WS), blk, 0, stream>>>(C1, C1t, XS, WS);
  k_transpose_cvt<<<g1((size_t)US * WS), blk, 0, stream>>>(D12, D12t, US, WS);
  k_transpose_cvt<<<g1((size_t)WS * WS), blk, 0, stream>>>(D11, D11t, WS, WS);
  k_transpose_cvt<<<g1((size_t)XS * XS), blk, 0, stream>>>(A, At, XS, XS);
  k_transpose_cvt<<<g1((size_t)WS * XS), blk, 0, stream>>>(B1, B1t, WS, XS);
  k_transpose_cvt<<<g1((size_t)US * XS), blk, 0, stream>>>(B2, B2t, US, XS);
  k_transpose_cvt<<<g1((size_t)XS * YS), blk, 0, stream>>>(C2, C2t, XS, YS);
  k_transpose_cvt<<<g1((size_t)WS * YS), blk, 0, stream>>>(D21, D21t, WS, YS);
  k_transpose_cvt<<<g1((size_t)US * YS), blk, 0, stream>>>(D22, D22t, US, YS);

  // 2) activations -> bf16
  k_cvt_bf16<<<g1((size_t)Bb * XS), blk, 0, stream>>>(x, xb, (size_t)Bb * XS);
  k_cvt_bf16<<<g1((size_t)Bb * US), blk, 0, stream>>>(u, ub, (size_t)Bb * US);

  // 3) b = x@C1 + u@D12   (f32)
  k_wmma_gemm<<<dim3(Bb / 128, WS / 128), blk, 0, stream>>>(
      xb, C1t, XS, ub, D12t, US, nullptr, nullptr, 0, nullptr, bmat, nullptr,
      Bb, WS, 0);

  // 4) w = 0; 30x  w = tanh(b + w@D11)
  k_zero_bf16<<<g1((size_t)Bb * WS), blk, 0, stream>>>(w0, (size_t)Bb * WS);
  __bf16* wc = w0;
  __bf16* wn = w1;
  for (int it = 0; it < NITER; ++it) {
    k_wmma_gemm<<<dim3(Bb / 128, WS / 128), blk, 0, stream>>>(
        wc, D11t, WS, nullptr, nullptr, 0, nullptr, nullptr, 0, bmat, nullptr,
        wn, Bb, WS, 1);
    __bf16* t = wc; wc = wn; wn = t;
  }

  // 5) x_next = x@A + w@B1 + u@B2
  float* x_next = (float*)d_out;
  k_wmma_gemm<<<dim3(Bb / 128, XS / 128), blk, 0, stream>>>(
      xb, At, XS, wc, B1t, WS, ub, B2t, US, nullptr, x_next, nullptr, Bb, XS, 0);

  // 6) y = x@C2 + w@D21 + u@D22
  float* y = (float*)d_out + (size_t)Bb * XS;
  k_wmma_gemm<<<dim3(Bb / 128, YS / 128), blk, 0, stream>>>(
      xb, C2t, XS, wc, D21t, WS, ub, D22t, US, nullptr, y, nullptr, Bb, YS, 0);
}
